// Quantizer1d_64570538328101
// MI455X (gfx1250) — matrix operand order repaired and checked by reading, not compiled, not run on hardware
//
#include <hip/hip_runtime.h>

// Problem constants (match reference)
#define B_ 8
#define C_ 1024
#define T_ 2048
#define H_ 16
#define M_ 1024
#define D_ 64
#define ASTRIDE 72   // padded LDS row stride in halves: 16B-aligned, bank-conflict-free

typedef __attribute__((ext_vector_type(16))) _Float16 v16h;
typedef __attribute__((ext_vector_type(8)))  _Float16 v8h;
typedef __attribute__((ext_vector_type(8)))  float    v8f;

__device__ __forceinline__ v16h cat8(v8h p, v8h q){
  return __builtin_shufflevector(p, q, 0,1,2,3,4,5,6,7,8,9,10,11,12,13,14,15);
}

// ---------------- kernel 1: zero histogram ----------------
__global__ void k_zero(unsigned* __restrict__ counts, int n){
  int i = blockIdx.x*blockDim.x + threadIdx.x;
  if (i < n) counts[i] = 0u;
}

// ---------------- kernel 2: L2-normalize codebooks -> f16 [h][m][d] ----------------
__global__ __launch_bounds__(32) void k_cbnorm(const float* __restrict__ cb,
                                               _Float16* __restrict__ cbh){
  const int row = blockIdx.x;          // h*M + m
  const int l = threadIdx.x;           // 32 lanes, each handles d=l and d=l+32
  const float* p = cb + (size_t)row*D_;
  float a = p[l], b = p[l+32];
  float s = a*a + b*b;
  #pragma unroll
  for (int off = 16; off > 0; off >>= 1) s += __shfl_xor(s, off, 32);
  float scale = 1.0f / fmaxf(sqrtf(s), 1e-12f);
  cbh[(size_t)row*D_ + l]      = (_Float16)(a*scale);
  cbh[(size_t)row*D_ + l + 32] = (_Float16)(b*scale);
}

// ---------------- kernel 3: main WMMA argmax ----------------
// grid = (T/128, H, B), block = 128 (4 waves). Each wave: 32 query rows x 1024 codes.
__global__ __launch_bounds__(128) void k_main(const float* __restrict__ x,
                                              const _Float16* __restrict__ cbh,
                                              int* __restrict__ idxi,
                                              float* __restrict__ idxf,
                                              unsigned* __restrict__ counts){
  __shared__ _Float16 lds[4][2][16*ASTRIDE];
  const int wave = threadIdx.x >> 5;
  const int lane = threadIdx.x & 31;
  const int b = blockIdx.z, h = blockIdx.y;
  const int t0 = blockIdx.x * 128 + wave * 32;
  const int r  = lane & 15;    // row within tile / column within C tile
  const int dh = lane >> 4;    // which half of the wave

  // Stage two 16-row A tiles: load x coalesced, L2-normalize rows, f16 to LDS.
  for (int tt = 0; tt < 2; ++tt){
    const int tbase = t0 + tt*16;
    const float* xp = x + ((size_t)b*C_ + h*D_ + dh*32)*T_ + tbase + r;
    float v[32]; float ss = 0.f;
    #pragma unroll
    for (int j = 0; j < 32; ++j){ float val = xp[(size_t)j*T_]; v[j] = val; ss += val*val; }
    ss += __shfl_xor(ss, 16, 32);                    // lane pair holds full row
    float scale = 1.0f / fmaxf(sqrtf(ss), 1e-12f);
    _Float16* dst = &lds[wave][tt][r*ASTRIDE + dh*32];
    #pragma unroll
    for (int j = 0; j < 32; j += 8){
      v8h hv;
      #pragma unroll
      for (int k = 0; k < 8; ++k) hv[k] = (_Float16)(v[j+k]*scale);
      *(v8h*)(dst + j) = hv;                         // 16B-aligned LDS store
    }
  }
  __syncthreads();

  // A fragments per ISA 16-bit A layout: lane row = lane&15, kb = (lane>>4)*8,
  // v16h elems 0..7 -> K = w*32+kb+0..7, elems 8..15 -> K = w*32+kb+16..23.
  const int kb = dh * 8;
  v16h a[2][2];
  #pragma unroll
  for (int tt = 0; tt < 2; ++tt)
    #pragma unroll
    for (int w = 0; w < 2; ++w){
      const _Float16* s = &lds[wave][tt][r*ASTRIDE + w*32 + kb];
      a[tt][w] = cat8(*(const v8h*)s, *(const v8h*)(s + 16));
    }

  float bestv[2][8]; int besti[2][8];
  #pragma unroll
  for (int tt = 0; tt < 2; ++tt)
    #pragma unroll
    for (int i = 0; i < 8; ++i){ bestv[tt][i] = -3.4e38f; besti[tt][i] = 0; }

  const v8f cz = {0.f,0.f,0.f,0.f,0.f,0.f,0.f,0.f};
  const _Float16* cbbase = cbh + (size_t)h*M_*D_;

  for (int nt = 0; nt < M_/16; ++nt){
    // B fragments per ISA 16-bit B layout: lane col = lane&15,
    // v16h elems 0..7 -> K = w*32+(lane>>4)*8+0..7, elems 8..15 -> +16; codebook is row-major [m][d].
    const _Float16* crow = cbbase + (size_t)(nt*16 + r)*D_ + dh*8;
    v16h b0 = cat8(*(const v8h*)(crow),      *(const v8h*)(crow + 16));
    v16h b1 = cat8(*(const v8h*)(crow + 32), *(const v8h*)(crow + 48));

    v8f acc0 = __builtin_amdgcn_wmma_f32_16x16x32_f16(false, a[0][0], false, b0, (short)0, cz,   false, false);
    acc0     = __builtin_amdgcn_wmma_f32_16x16x32_f16(false, a[0][1], false, b1, (short)0, acc0, false, false);
    v8f acc1 = __builtin_amdgcn_wmma_f32_16x16x32_f16(false, a[1][0], false, b0, (short)0, cz,   false, false);
    acc1     = __builtin_amdgcn_wmma_f32_16x16x32_f16(false, a[1][1], false, b1, (short)0, acc1, false, false);

    const int col = nt*16 + r;
    #pragma unroll
    for (int i = 0; i < 8; ++i){
      float v0 = acc0[i]; if (v0 > bestv[0][i]) { bestv[0][i] = v0; besti[0][i] = col; }
      float v1 = acc1[i]; if (v1 > bestv[1][i]) { bestv[1][i] = v1; besti[1][i] = col; }
    }
  }

  // Reduce over the 16 columns held across each 16-lane group; tie -> lower index
  // (matches jnp.argmax first-occurrence semantics).
  #pragma unroll
  for (int tt = 0; tt < 2; ++tt){
    #pragma unroll
    for (int i = 0; i < 8; ++i){
      float v = bestv[tt][i]; int ix = besti[tt][i];
      #pragma unroll
      for (int mm = 1; mm < 16; mm <<= 1){
        float ov = __shfl_xor(v,  mm, 32);
        int   oi = __shfl_xor(ix, mm, 32);
        if (ov > v || (ov == v && oi < ix)) { v = ov; ix = oi; }
      }
      bestv[tt][i] = v; besti[tt][i] = ix;
    }
    if (r == 0){ // lane 0 -> rows 0..7, lane 16 -> rows 8..15 (C layout)
      const int tbase = t0 + tt*16 + dh*8;
      #pragma unroll
      for (int i = 0; i < 8; ++i){
        int m = besti[tt][i];
        size_t io = ((size_t)b*H_ + h)*T_ + tbase + i;
        idxi[io] = m;
        idxf[io] = (float)m;
        atomicAdd(&counts[h*M_ + m], 1u);
      }
    }
  }
}

// ---------------- kernel 4: gather out[b, h*64+d, t] = cb[h, idx[b,h,t], d] ----------------
__global__ void k_gather(const float* __restrict__ cb, const int* __restrict__ idx,
                         float* __restrict__ out){
  size_t i = (size_t)blockIdx.x*256 + threadIdx.x;   // flat over B*C*T, coalesced in t
  int t = (int)(i & (T_-1));
  int c = (int)((i >> 11) & (C_-1));
  int b = (int)(i >> 21);
  int h = c >> 6, d = c & 63;
  int m = idx[((size_t)b*H_ + h)*T_ + t];
  out[i] = cb[((size_t)h*M_ + m)*D_ + d];
}

// ---------------- kernel 5: perplexity per head ----------------
__global__ __launch_bounds__(256) void k_perp(const unsigned* __restrict__ counts,
                                              float* __restrict__ perp){
  const int h = blockIdx.x, tid = threadIdx.x;
  const float inv = 1.0f / (float)(B_*T_);
  float s = 0.f;
  for (int m = tid; m < M_; m += 256){
    float p = (float)counts[h*M_ + m] * inv;
    s += p * logf(p + 1e-10f);
  }
  __shared__ float red[256];
  red[tid] = s; __syncthreads();
  for (int k = 128; k > 0; k >>= 1){ if (tid < k) red[tid] += red[tid+k]; __syncthreads(); }
  if (tid == 0) perp[h] = expf(-red[0]);
}

extern "C" void kernel_launch(void* const* d_in, const int* in_sizes, int n_in,
                              void* d_out, int out_size, void* d_ws, size_t ws_size,
                              hipStream_t stream) {
  const float* x  = (const float*)d_in[0];   // (B, C, T) f32
  const float* cb = (const float*)d_in[1];   // (H, M, D) f32

  float* out      = (float*)d_out;                       // (B, C, T)
  float* out_idx  = out + (size_t)B_*C_*T_;              // (B, H, T) indices as float
  float* out_perp = out_idx + (size_t)B_*H_*T_;          // (H,)

  char* ws = (char*)d_ws;
  _Float16* cbh   = (_Float16*)ws;                                  // 2 MB normalized f16 codebooks
  unsigned* cnts  = (unsigned*)(ws + (size_t)2*1024*1024);          // 64 KB histogram
  int*      idxi  = (int*)(ws + (size_t)2*1024*1024 + 64*1024);     // 1 MB int indices

  k_zero<<<(H_*M_ + 255)/256, 256, 0, stream>>>(cnts, H_*M_);
  k_cbnorm<<<H_*M_, 32, 0, stream>>>(cb, cbh);

  dim3 g(T_/128, H_, B_);
  k_main<<<g, 128, 0, stream>>>(x, cbh, idxi, out_idx, cnts);

  k_gather<<<(int)(((size_t)B_*C_*T_)/256), 256, 0, stream>>>(cb, idxi, out);
  k_perp<<<H_, 256, 0, stream>>>(cnts, out_perp);
}
